// MLPQuadrupedProjectionFilter_52725018525889
// MI455X (gfx1250) — matrix operand order repaired and checked by reading, not compiled, not run on hardware
//
#include <hip/hip_runtime.h>
#include <hip/hip_bf16.h>
#include <math.h>

typedef __attribute__((ext_vector_type(16))) _Float16 v16h;
typedef __attribute__((ext_vector_type(8)))  _Float16 v8h;
typedef __attribute__((ext_vector_type(8)))  float    v8f;

#define NVARQ   600
#define NCQ     1000
#define NEQQ    150
#define HIDQ    1024
#define GINQ    4800
#define GOUTQ   1600
#define MLPOUTQ 2200
#define KKTQ    750
#define BATCHQ  1024
#define MAXITQ  15
#define BEQZ    294.3f   /* 30.0 * 9.81 */

// padded f16 strides (multiples of 32 halves; pad columns are written as zero)
#define S600    608
#define S750    768
#define S1000   1024

// ---------------------------------------------------------------------------
// Async global->LDS copy of 32 contiguous bytes (2 x b128).  The instruction
// adds the same immediate offset to both the LDS (VDST vgpr) and global
// (VADDR vgpr pair) addresses, so one base pair covers both halves.
// Tracked by ASYNCcnt (gfx1250 async-tensor path).
// ---------------------------------------------------------------------------
__device__ __forceinline__ void async_copy32(unsigned lds, const _Float16* g)
{
    asm volatile("global_load_async_to_lds_b128 %0, %1, off"
                 :: "v"(lds), "v"(g) : "memory");
    asm volatile("global_load_async_to_lds_b128 %0, %1, off offset:16"
                 :: "v"(lds), "v"(g) : "memory");
}

__device__ __forceinline__ void wait_async0()
{
    asm volatile("s_wait_asynccnt 0x0" ::: "memory");
}

// ---------------------------------------------------------------------------
// WMMA GEMM:  D = act(alpha * (A @ B^T) + beta*Cin + bias)
// A: (M x Kp) f16 row-major stride lda (pad cols zero), B: (N x Kp) stride ldb.
// Kp (= lda = ldb in all uses here) is a multiple of 32 -> no K guards.
// Row indices are clamped (branch-free); garbage rows only feed accumulator
// lanes whose stores are guarded off.  Block tile 128x128, 8 waves (2x4),
// wave tile 64x32 -> 4x2 v_wmma_f32_16x16x32_f16 per K step.
// Slabs move global->LDS via async copies, double-buffered: one barrier per
// K-step, copies for step i+1 overlap the WMMAs of step i.
// ---------------------------------------------------------------------------
__global__ __launch_bounds__(256)
void k_gemm(const _Float16* __restrict__ A, int lda,
            const _Float16* __restrict__ B, int ldb,
            float* __restrict__ Dout, int ldd,
            _Float16* __restrict__ Dh, int ldh,
            const float* __restrict__ bias,
            const float* __restrict__ Cin, int ldc,
            float alpha, float beta, int act,
            int M, int N, int Kp)
{
    __shared__ _Float16 As[2][128][40];   // +8 half pad per row
    __shared__ _Float16 Bs[2][128][40];

    const int tid  = threadIdx.x;
    const int lane = tid & 31;
    const int wave = tid >> 5;
    const int wm   = wave >> 2;        // 0..1
    const int wn   = wave & 3;         // 0..3
    const int l16  = lane & 15;
    const int lhi  = lane >> 4;        // 0/1
    const int rowBase = blockIdx.y * 128;
    const int colBase = blockIdx.x * 128;

    v8f acc[4][2] = {};

    const int ldRow = tid >> 1;        // 0..127
    const int ldOff = (tid & 1) * 16;  // 0 or 16

    union HU { v16h v; v8h h[2]; };

    // branch-free row clamp; OOB rows feed only never-stored output rows
    const int gra = (rowBase + ldRow < M) ? (rowBase + ldRow) : (M - 1);
    const int gcb = (colBase + ldRow < N) ? (colBase + ldRow) : (N - 1);
    const _Float16* pa = A + (size_t)gra * lda + ldOff;
    const _Float16* pb = B + (size_t)gcb * ldb + ldOff;

    // per-lane LDS byte offsets (low 32 bits of the flat shared address)
    unsigned ldsA0 = (unsigned)(size_t)&As[0][ldRow][ldOff];
    unsigned ldsA1 = (unsigned)(size_t)&As[1][ldRow][ldOff];
    unsigned ldsB0 = (unsigned)(size_t)&Bs[0][ldRow][ldOff];
    unsigned ldsB1 = (unsigned)(size_t)&Bs[1][ldRow][ldOff];

    // prologue: stage k0 = 0 into buffer 0
    async_copy32(ldsA0, pa);
    async_copy32(ldsB0, pb);

    int buf = 0;
    for (int k0 = 0; k0 < Kp; k0 += 32) {
        wait_async0();        // own slab landed in LDS[buf]
        __syncthreads();      // everyone's slab landed; prev buffer free

        if (k0 + 32 < Kp) {   // overlap next slab copy with the WMMAs
            async_copy32(buf ? ldsA0 : ldsA1, pa + k0 + 32);
            async_copy32(buf ? ldsB0 : ldsB1, pb + k0 + 32);
        }

        // fragments per ISA lane layout
        // A 16x32: lane<16 -> K {0..7},{16..23}; lane>=16 -> K {8..15},{24..31}
        v16h afr[4];
        #pragma unroll
        for (int mi = 0; mi < 4; ++mi) {
            const _Float16* sp = &As[buf][wm * 64 + mi * 16 + l16][lhi * 8];
            HU u; u.h[0] = *(const v8h*)sp; u.h[1] = *(const v8h*)(sp + 16);
            afr[mi] = u.v;
        }
        // B 32x16: lane<16 -> col, K {0..7},{16..23}; lane>=16 -> col, K {8..15},{24..31}
        v16h bfr[2];
        #pragma unroll
        for (int ni = 0; ni < 2; ++ni) {
            const _Float16* sp = &Bs[buf][wn * 32 + ni * 16 + l16][lhi * 8];
            HU u; u.h[0] = *(const v8h*)sp; u.h[1] = *(const v8h*)(sp + 16);
            bfr[ni] = u.v;
        }

        #pragma unroll
        for (int mi = 0; mi < 4; ++mi)
            #pragma unroll
            for (int ni = 0; ni < 2; ++ni)
                acc[mi][ni] = __builtin_amdgcn_wmma_f32_16x16x32_f16(
                    false, afr[mi], false, bfr[ni],
                    (short)0, acc[mi][ni], false, false);

        buf ^= 1;
    }

    // epilogue: C/D layout: VGPR v -> row = v + 8*(lane>=16), col = lane&15
    #pragma unroll
    for (int mi = 0; mi < 4; ++mi) {
        int r0 = rowBase + wm * 64 + mi * 16 + lhi * 8;
        #pragma unroll
        for (int ni = 0; ni < 2; ++ni) {
            int cc = colBase + wn * 32 + ni * 16 + l16;
            if (cc < N) {
                #pragma unroll
                for (int v = 0; v < 8; ++v) {
                    int r = r0 + v;
                    if (r < M) {
                        float val = alpha * acc[mi][ni][v];
                        if (Cin)  val += beta * Cin[(size_t)r * ldc + cc];
                        if (bias) val += bias[cc];
                        if (act == 1)      val = fmaxf(val, 0.0f);
                        else if (act == 2) val = tanhf(val);
                        if (Dout) Dout[(size_t)r * ldd + cc] = val;
                        if (Dh)   Dh[(size_t)r * ldh + cc] = (_Float16)val;
                    }
                }
            }
        }
    }
}

// ---------------------------------------------------------------------------
// Stats: deterministic two-pass mean/std (ddof=1) over whole x
// ---------------------------------------------------------------------------
__global__ __launch_bounds__(256)
void k_reduce_stats(const float* __restrict__ x, int n, float* __restrict__ partials)
{
    __shared__ float ss[256], sq[256];
    float s = 0.f, q = 0.f;
    for (int i = blockIdx.x * blockDim.x + threadIdx.x; i < n;
         i += gridDim.x * blockDim.x) {
        float v = x[i]; s += v; q += v * v;
    }
    ss[threadIdx.x] = s; sq[threadIdx.x] = q; __syncthreads();
    for (int o = 128; o > 0; o >>= 1) {
        if (threadIdx.x < o) { ss[threadIdx.x] += ss[threadIdx.x + o];
                               sq[threadIdx.x] += sq[threadIdx.x + o]; }
        __syncthreads();
    }
    if (threadIdx.x == 0) {
        partials[blockIdx.x * 2]     = ss[0];
        partials[blockIdx.x * 2 + 1] = sq[0];
    }
}

__global__ void k_finalize_stats(const float* partials, int nb, int n, float* stats)
{
    if (threadIdx.x == 0 && blockIdx.x == 0) {
        double s = 0.0, q = 0.0;
        for (int i = 0; i < nb; ++i) { s += partials[i * 2]; q += partials[i * 2 + 1]; }
        float mean = (float)(s / n);
        float var  = (float)((q - s * s / n) / (n - 1));
        float sd   = sqrtf(fmaxf(var, 0.f));
        stats[0] = mean;
        stats[1] = 1.f / (sd + 1e-8f);
    }
}

// normalize + cast to f16, padded stride S600 (pad cols zero)
__global__ void k_norm_cast(const float* __restrict__ x, const float* __restrict__ stats,
                            _Float16* __restrict__ out)
{
    int i = blockIdx.x * 256 + threadIdx.x;
    if (i < BATCHQ * S600) {
        int m = i / S600, j = i % S600;
        out[i] = (j < NVARQ) ? (_Float16)((x[m * NVARQ + j] - stats[0]) * stats[1])
                             : (_Float16)0.0f;
    }
}

// generic f32 -> f16 cast with zero-padded destination stride
__global__ void k_cast(const float* __restrict__ src, _Float16* __restrict__ dst,
                       int rows, int w, int dstride)
{
    long long i = (long long)blockIdx.x * 256 + threadIdx.x;
    long long total = (long long)rows * dstride;
    if (i < total) {
        int r = (int)(i / dstride), c = (int)(i % dstride);
        dst[i] = (c < w) ? (_Float16)src[(long long)r * w + c] : (_Float16)0.0f;
    }
}

// C (NC x NVAR) -> CT f16 (NVAR x S1000), pad zero
__global__ void k_transpose_cast(const float* __restrict__ C, _Float16* __restrict__ CT)
{
    int i = blockIdx.x * 256 + threadIdx.x;
    if (i < NVARQ * S1000) {
        int n = i / S1000, k = i % S1000;
        CT[i] = (k < NCQ) ? (_Float16)C[k * NVARQ + n] : (_Float16)0.0f;
    }
}

// ---------------------------------------------------------------------------
// KKT assembly + single-workgroup Gauss-Jordan inverse (750x750)
// ---------------------------------------------------------------------------
__global__ void k_assemble_kkt(const float* __restrict__ costCC,
                               const float* __restrict__ H, float* __restrict__ aug)
{
    int i = blockIdx.x * 256 + threadIdx.x;
    if (i >= KKTQ * 2 * KKTQ) return;
    int r = i / (2 * KKTQ), c = i % (2 * KKTQ);
    float v = 0.f;
    if (c < KKTQ) {
        if (r < NVARQ && c < NVARQ)
            v = costCC[r * NVARQ + c] + H[r * NVARQ + c];
        else if (r < NVARQ) {            // A_eq^T block
            int e = c - NVARQ;
            v = ((r / 12) == (e / 3) && (r % 3) == (e % 3)) ? 1.f : 0.f;
        } else if (c < NVARQ) {          // A_eq block
            int e = r - NVARQ;
            v = ((c / 12) == (e / 3) && (c % 3) == (e % 3)) ? 1.f : 0.f;
        }
    } else {
        v = ((c - KKTQ) == r) ? 1.f : 0.f;
    }
    aug[r * 2 * KKTQ + c] = v;
}

__global__ __launch_bounds__(1024)
void k_gauss_jordan(float* __restrict__ aug)
{
    __shared__ float prow[2 * KKTQ];
    __shared__ float fac[KKTQ];
    const int tid = threadIdx.x;
    for (int p = 0; p < KKTQ; ++p) {
        float piv = aug[p * 2 * KKTQ + p];
        float inv = 1.f / piv;
        __syncthreads();
        for (int j = tid; j < 2 * KKTQ; j += 1024) {
            float v = aug[p * 2 * KKTQ + j] * inv;
            aug[p * 2 * KKTQ + j] = v;
            prow[j] = v;
        }
        __syncthreads();
        for (int i = tid; i < KKTQ; i += 1024) fac[i] = aug[i * 2 * KKTQ + p];
        __syncthreads();
        for (int i = 0; i < KKTQ; ++i) {
            if (i == p) continue;
            float f = fac[i];
            for (int j = tid; j < 2 * KKTQ; j += 1024)
                aug[i * 2 * KKTQ + j] -= f * prow[j];
        }
        __syncthreads();
    }
}

// Qinv (right half of aug) -> f16, stride S750, pad zero
__global__ void k_qinv_cast(const float* __restrict__ aug, _Float16* __restrict__ q)
{
    int i = blockIdx.x * 256 + threadIdx.x;
    if (i < KKTQ * S750) {
        int r = i / S750, c = i % S750;
        q[i] = (c < KKTQ) ? (_Float16)aug[r * 2 * KKTQ + KKTQ + c] : (_Float16)0.0f;
    }
}

// ---------------------------------------------------------------------------
// Elementwise / reduction kernels for the iteration loop
// ---------------------------------------------------------------------------
__global__ void k_split_nn(const float* __restrict__ nn, float* __restrict__ lam,
                           float* __restrict__ s)
{
    int i = blockIdx.x * 256 + threadIdx.x;
    if (i < BATCHQ * (NVARQ + NCQ)) {
        int m = i / (NVARQ + NCQ), j = i % (NVARQ + NCQ);
        if (j < NVARQ) lam[m * NVARQ + j] = nn[m * MLPOUTQ + NVARQ + j];
        else {
            int t = j - NVARQ;
            s[m * NCQ + t] = fmaxf(0.f, nn[m * MLPOUTQ + 2 * NVARQ + t]);
        }
    }
}

// baug = c - s  (f16, stride S1000, pad zero)
__global__ void k_baug(const float* __restrict__ s, const float* __restrict__ c,
                       _Float16* __restrict__ out)
{
    int i = blockIdx.x * 256 + threadIdx.x;
    if (i < BATCHQ * S1000) {
        int m = i >> 10, j = i & (S1000 - 1);
        out[i] = (j < NCQ) ? (_Float16)(c[j] - s[m * NCQ + j]) : (_Float16)0.0f;
    }
}

// rhs = [lam - g + tmpA , b_eq] (f16, stride S750, pad zero)
__global__ void k_build_rhs(const float* __restrict__ tmpA, const float* __restrict__ lam,
                            const float* __restrict__ g, _Float16* __restrict__ rhs)
{
    int i = blockIdx.x * 256 + threadIdx.x;
    if (i < BATCHQ * S750) {
        int m = i / S750, j = i % S750;
        float v = 0.f;
        if (j < NVARQ)      v = lam[m * NVARQ + j] - g[j] + tmpA[m * NVARQ + j];
        else if (j < KKTQ)  v = (((j - NVARQ) % 3) == 2) ? BEQZ : 0.f;
        rhs[i] = (_Float16)v;
    }
}

// solh = f16(sol[:, :NVAR]) with stride S600, pad zero
__global__ void k_solh_cast(const float* __restrict__ sol, _Float16* __restrict__ solh)
{
    int i = blockIdx.x * 256 + threadIdx.x;
    if (i < BATCHQ * S600) {
        int m = i / S600, j = i % S600;
        solh[i] = (j < NVARQ) ? (_Float16)sol[m * KKTQ + j] : (_Float16)0.0f;
    }
}

__global__ __launch_bounds__(256)
void k_post_cx(const float* __restrict__ Cx, const float* __restrict__ c,
               float* __restrict__ sN, _Float16* __restrict__ resh,
               float* __restrict__ primal)
{
    __shared__ float red[256];
    int row = blockIdx.x;
    float acc = 0.f;
    for (int j = threadIdx.x; j < S1000; j += 256) {
        if (j < NCQ) {
            float cx = Cx[row * NCQ + j];
            float sn = fmaxf(0.f, c[j] - cx);
            float rv = cx - c[j] + sn;
            sN[row * NCQ + j] = sn;
            resh[row * S1000 + j] = (_Float16)rv;
            acc += rv * rv;
        } else {
            resh[row * S1000 + j] = (_Float16)0.0f;   // pad for GEMM
        }
    }
    red[threadIdx.x] = acc; __syncthreads();
    for (int o = 128; o > 0; o >>= 1) {
        if (threadIdx.x < o) red[threadIdx.x] += red[threadIdx.x + o];
        __syncthreads();
    }
    if (threadIdx.x == 0) primal[row] = sqrtf(red[0]);
}

__global__ void k_build_r(const float* __restrict__ s, const float* __restrict__ lam,
                          const float* __restrict__ sN, const float* __restrict__ lamN,
                          _Float16* __restrict__ r)
{
    long long i = (long long)blockIdx.x * 256 + threadIdx.x;
    if (i < (long long)BATCHQ * GINQ) {
        int m = (int)(i / GINQ), j = (int)(i % GINQ);
        float v;
        if (j < NCQ)                      v = s[m * NCQ + j];
        else if (j < NCQ + NVARQ)         v = lam[m * NVARQ + (j - NCQ)];
        else if (j < 2 * NCQ + NVARQ)     v = sN[m * NCQ + (j - NCQ - NVARQ)];
        else if (j < 2 * NCQ + 2 * NVARQ) v = lamN[m * NVARQ + (j - 2 * NCQ - NVARQ)];
        else if (j < 3 * NCQ + 2 * NVARQ) {
            int t = j - 2 * NCQ - 2 * NVARQ;
            v = sN[m * NCQ + t] - s[m * NCQ + t];
        } else {
            int t = j - 3 * NCQ - 2 * NVARQ;
            v = lamN[m * NVARQ + t] - lam[m * NVARQ + t];
        }
        r[i] = (_Float16)v;
    }
}

__global__ void k_gru(const float* __restrict__ gi, const float* __restrict__ gh,
                      float* __restrict__ h, _Float16* __restrict__ hh)
{
    int i = blockIdx.x * 256 + threadIdx.x;
    if (i < BATCHQ * HIDQ) {
        int m = i >> 10, j = i & 1023;
        const float* gim = gi + (size_t)m * 3 * HIDQ;
        const float* ghm = gh + (size_t)m * 3 * HIDQ;
        float rg = 1.f / (1.f + __expf(-(gim[j] + ghm[j])));
        float z  = 1.f / (1.f + __expf(-(gim[HIDQ + j] + ghm[HIDQ + j])));
        float n  = tanhf(gim[2 * HIDQ + j] + rg * ghm[2 * HIDQ + j]);
        float hn = (1.f - z) * n + z * h[i];
        h[i] = hn;
        hh[i] = (_Float16)hn;
    }
}

__global__ __launch_bounds__(256)
void k_finalize(const float* __restrict__ gout, float* __restrict__ s,
                const float* __restrict__ sN, float* __restrict__ lam,
                const float* __restrict__ lamN, float* __restrict__ fp)
{
    __shared__ float red[256];
    int row = blockIdx.x;
    float a = 0.f;
    for (int j = threadIdx.x; j < NCQ; j += 256) {
        float sf = fmaxf(0.f, sN[row * NCQ + j] + gout[row * GOUTQ + j]);
        float d  = sf - s[row * NCQ + j];
        a += d * d;
        s[row * NCQ + j] = sf;
    }
    red[threadIdx.x] = a; __syncthreads();
    for (int o = 128; o > 0; o >>= 1) {
        if (threadIdx.x < o) red[threadIdx.x] += red[threadIdx.x + o];
        __syncthreads();
    }
    float fa = sqrtf(red[0]);
    __syncthreads();
    float b = 0.f;
    for (int j = threadIdx.x; j < NVARQ; j += 256) {
        float lf = lamN[row * NVARQ + j] + gout[row * GOUTQ + NCQ + j];
        float d  = lf - lam[row * NVARQ + j];
        b += d * d;
        lam[row * NVARQ + j] = lf;
    }
    red[threadIdx.x] = b; __syncthreads();
    for (int o = 128; o > 0; o >>= 1) {
        if (threadIdx.x < o) red[threadIdx.x] += red[threadIdx.x + o];
        __syncthreads();
    }
    if (threadIdx.x == 0) fp[row] = fa + sqrtf(red[0]);
}

__global__ void k_copy_xi(const float* __restrict__ sol, float* __restrict__ out)
{
    int i = blockIdx.x * 256 + threadIdx.x;
    if (i < BATCHQ * NVARQ) {
        int m = i / NVARQ, j = i % NVARQ;
        out[i] = sol[m * KKTQ + j];
    }
}

__global__ void k_avg(const float* __restrict__ fp_hist, const float* __restrict__ pr_hist,
                      float* __restrict__ avg_fp, float* __restrict__ avg_pr)
{
    int i = blockIdx.x * 256 + threadIdx.x;
    if (i < BATCHQ) {
        float a = 0.f, b = 0.f;
        for (int t = 0; t < MAXITQ; ++t) {
            a += fp_hist[t * BATCHQ + i];
            b += pr_hist[t * BATCHQ + i];
        }
        avg_fp[i] = a / MAXITQ;
        avg_pr[i] = b / MAXITQ;
    }
}

// ---------------------------------------------------------------------------
static inline void launch_gemm(hipStream_t st,
    const _Float16* A, int lda, const _Float16* B, int ldb,
    float* Dout, int ldd, _Float16* Dh, int ldh,
    const float* bias, const float* Cin, int ldc,
    float alpha, float beta, int act, int M, int N, int Kp)
{
    dim3 grid((N + 127) / 128, (M + 127) / 128);
    k_gemm<<<grid, 256, 0, st>>>(A, lda, B, ldb, Dout, ldd, Dh, ldh,
                                 bias, Cin, ldc, alpha, beta, act, M, N, Kp);
}

static inline int gb(long long n) { return (int)((n + 255) / 256); }

extern "C" void kernel_launch(void* const* d_in, const int* in_sizes, int n_in,
                              void* d_out, int out_size, void* d_ws, size_t ws_size,
                              hipStream_t stream)
{
    (void)in_sizes; (void)n_in; (void)out_size; (void)ws_size;
    const float* x     = (const float*)d_in[0];
    const float* H     = (const float*)d_in[1];
    const float* g     = (const float*)d_in[2];
    const float* C     = (const float*)d_in[3];
    const float* c     = (const float*)d_in[4];
    const float* W1    = (const float*)d_in[5];
    const float* b1    = (const float*)d_in[6];
    const float* W2    = (const float*)d_in[7];
    const float* b2    = (const float*)d_in[8];
    const float* W3    = (const float*)d_in[9];
    const float* b3    = (const float*)d_in[10];
    const float* Wg    = (const float*)d_in[11];
    const float* bg    = (const float*)d_in[12];
    const float* W_ih  = (const float*)d_in[13];
    const float* W_hh  = (const float*)d_in[14];
    const float* b_ih  = (const float*)d_in[15];
    const float* b_hh  = (const float*)d_in[16];
    const float* W_out = (const float*)d_in[17];
    const float* b_out = (const float*)d_in[18];

    float* out        = (float*)d_out;
    float* out_xi     = out;                       // 1024*600
    float* out_avg_fp = out + 614400;              // 1024
    float* out_avg_pr = out + 615424;              // 1024
    float* out_primal = out + 616448;              // 15*1024
    float* out_fp     = out + 631808;              // 15*1024

    char* wp = (char*)d_ws;
    auto alloc = [&](size_t bytes) -> void* {
        void* p = (void*)wp;
        wp += (bytes + 255) & ~(size_t)255;
        return p;
    };

    float*    partials = (float*)alloc(256 * 2 * sizeof(float));
    float*    stats    = (float*)alloc(16);
    _Float16* normh  = (_Float16*)alloc((size_t)BATCHQ * S600 * 2);
    _Float16* W1h    = (_Float16*)alloc((size_t)HIDQ * S600 * 2);
    _Float16* W2h    = (_Float16*)alloc((size_t)HIDQ * HIDQ * 2);
    _Float16* W3h    = (_Float16*)alloc((size_t)MLPOUTQ * HIDQ * 2);
    _Float16* Wgh    = (_Float16*)alloc((size_t)HIDQ * S600 * 2);
    _Float16* Ch     = (_Float16*)alloc((size_t)NCQ * S600 * 2);
    _Float16* CTh    = (_Float16*)alloc((size_t)NVARQ * S1000 * 2);
    _Float16* Wihh   = (_Float16*)alloc((size_t)3 * HIDQ * GINQ * 2);
    _Float16* Whhh   = (_Float16*)alloc((size_t)3 * HIDQ * HIDQ * 2);
    _Float16* Wouth  = (_Float16*)alloc((size_t)GOUTQ * HIDQ * 2);
    _Float16* Qinvh  = (_Float16*)alloc((size_t)KKTQ * S750 * 2);
    float*    costCC = (float*)alloc((size_t)NVARQ * NVARQ * 4);
    float*    aug    = (float*)alloc((size_t)KKTQ * 2 * KKTQ * 4);
    _Float16* h1h    = (_Float16*)alloc((size_t)BATCHQ * HIDQ * 2);
    _Float16* h2h    = (_Float16*)alloc((size_t)BATCHQ * HIDQ * 2);
    float*    nn_out = (float*)alloc((size_t)BATCHQ * MLPOUTQ * 4);
    float*    hbuf   = (float*)alloc((size_t)BATCHQ * HIDQ * 4);
    _Float16* hh     = (_Float16*)alloc((size_t)BATCHQ * HIDQ * 2);
    float*    s      = (float*)alloc((size_t)BATCHQ * NCQ * 4);
    float*    sN     = (float*)alloc((size_t)BATCHQ * NCQ * 4);
    float*    lam    = (float*)alloc((size_t)BATCHQ * NVARQ * 4);
    float*    lamN   = (float*)alloc((size_t)BATCHQ * NVARQ * 4);
    _Float16* baugh  = (_Float16*)alloc((size_t)BATCHQ * S1000 * 2);
    float*    tmpA   = (float*)alloc((size_t)BATCHQ * NVARQ * 4);
    _Float16* rhsh   = (_Float16*)alloc((size_t)BATCHQ * S750 * 2);
    float*    sol    = (float*)alloc((size_t)BATCHQ * KKTQ * 4);
    _Float16* solh   = (_Float16*)alloc((size_t)BATCHQ * S600 * 2);
    float*    Cx     = (float*)alloc((size_t)BATCHQ * NCQ * 4);
    _Float16* resh   = (_Float16*)alloc((size_t)BATCHQ * S1000 * 2);
    _Float16* rbuf   = (_Float16*)alloc((size_t)BATCHQ * GINQ * 2);
    float*    gi     = (float*)alloc((size_t)BATCHQ * 3 * HIDQ * 4);
    float*    ghb    = (float*)alloc((size_t)BATCHQ * 3 * HIDQ * 4);
    float*    gout   = (float*)alloc((size_t)BATCHQ * GOUTQ * 4);

    // ---- prologue: stats + normalize + weight casts (pad cols zeroed) ----
    k_reduce_stats<<<256, 256, 0, stream>>>(x, BATCHQ * NVARQ, partials);
    k_finalize_stats<<<1, 1, 0, stream>>>(partials, 256, BATCHQ * NVARQ, stats);
    k_norm_cast<<<gb((long long)BATCHQ * S600), 256, 0, stream>>>(x, stats, normh);

    k_cast<<<gb((long long)HIDQ * S600), 256, 0, stream>>>(W1, W1h, HIDQ, NVARQ, S600);
    k_cast<<<gb((long long)HIDQ * HIDQ), 256, 0, stream>>>(W2, W2h, HIDQ, HIDQ, HIDQ);
    k_cast<<<gb((long long)MLPOUTQ * HIDQ), 256, 0, stream>>>(W3, W3h, MLPOUTQ, HIDQ, HIDQ);
    k_cast<<<gb((long long)HIDQ * S600), 256, 0, stream>>>(Wg, Wgh, HIDQ, NVARQ, S600);
    k_cast<<<gb((long long)NCQ * S600), 256, 0, stream>>>(C, Ch, NCQ, NVARQ, S600);
    k_cast<<<gb((long long)3 * HIDQ * GINQ), 256, 0, stream>>>(W_ih, Wihh, 3 * HIDQ, GINQ, GINQ);
    k_cast<<<gb((long long)3 * HIDQ * HIDQ), 256, 0, stream>>>(W_hh, Whhh, 3 * HIDQ, HIDQ, HIDQ);
    k_cast<<<gb((long long)GOUTQ * HIDQ), 256, 0, stream>>>(W_out, Wouth, GOUTQ, HIDQ, HIDQ);
    k_transpose_cast<<<gb((long long)NVARQ * S1000), 256, 0, stream>>>(C, CTh);

    // ---- KKT: cost = H + C^T C ; invert via Gauss-Jordan ----
    launch_gemm(stream, CTh, S1000, CTh, S1000, costCC, NVARQ, nullptr, 0,
                nullptr, nullptr, 0, 1.f, 0.f, 0, NVARQ, NVARQ, S1000);
    k_assemble_kkt<<<gb((long long)KKTQ * 2 * KKTQ), 256, 0, stream>>>(costCC, H, aug);
    k_gauss_jordan<<<1, 1024, 0, stream>>>(aug);
    k_qinv_cast<<<gb((long long)KKTQ * S750), 256, 0, stream>>>(aug, Qinvh);

    // ---- MLP + GRU init ----
    launch_gemm(stream, normh, S600, W1h, S600, nullptr, 0, h1h, HIDQ,
                b1, nullptr, 0, 1.f, 0.f, 1, BATCHQ, HIDQ, S600);
    launch_gemm(stream, h1h, HIDQ, W2h, HIDQ, nullptr, 0, h2h, HIDQ,
                b2, nullptr, 0, 1.f, 0.f, 1, BATCHQ, HIDQ, HIDQ);
    launch_gemm(stream, h2h, HIDQ, W3h, HIDQ, nn_out, MLPOUTQ, nullptr, 0,
                b3, nullptr, 0, 1.f, 0.f, 0, BATCHQ, MLPOUTQ, HIDQ);
    launch_gemm(stream, normh, S600, Wgh, S600, hbuf, HIDQ, hh, HIDQ,
                bg, nullptr, 0, 1.f, 0.f, 2, BATCHQ, HIDQ, S600);
    k_split_nn<<<gb((long long)BATCHQ * (NVARQ + NCQ)), 256, 0, stream>>>(nn_out, lam, s);

    // ---- ADMM iterations ----
    for (int it = 0; it < MAXITQ; ++it) {
        k_baug<<<gb((long long)BATCHQ * S1000), 256, 0, stream>>>(s, c, baugh);
        launch_gemm(stream, baugh, S1000, CTh, S1000, tmpA, NVARQ, nullptr, 0,
                    nullptr, nullptr, 0, 1.f, 0.f, 0, BATCHQ, NVARQ, S1000);
        k_build_rhs<<<gb((long long)BATCHQ * S750), 256, 0, stream>>>(tmpA, lam, g, rhsh);
        launch_gemm(stream, rhsh, S750, Qinvh, S750, sol, KKTQ, nullptr, 0,
                    nullptr, nullptr, 0, 1.f, 0.f, 0, BATCHQ, KKTQ, S750);
        k_solh_cast<<<gb((long long)BATCHQ * S600), 256, 0, stream>>>(sol, solh);
        launch_gemm(stream, solh, S600, Ch, S600, Cx, NCQ, nullptr, 0,
                    nullptr, nullptr, 0, 1.f, 0.f, 0, BATCHQ, NCQ, S600);
        k_post_cx<<<BATCHQ, 256, 0, stream>>>(Cx, c, sN, resh, out_primal + it * BATCHQ);
        launch_gemm(stream, resh, S1000, CTh, S1000, lamN, NVARQ, nullptr, 0,
                    nullptr, lam, NVARQ, -1.f, 1.f, 0, BATCHQ, NVARQ, S1000);
        k_build_r<<<gb((long long)BATCHQ * GINQ), 256, 0, stream>>>(s, lam, sN, lamN, rbuf);
        launch_gemm(stream, rbuf, GINQ, Wihh, GINQ, gi, 3 * HIDQ, nullptr, 0,
                    b_ih, nullptr, 0, 1.f, 0.f, 0, BATCHQ, 3 * HIDQ, GINQ);
        launch_gemm(stream, hh, HIDQ, Whhh, HIDQ, ghb, 3 * HIDQ, nullptr, 0,
                    b_hh, nullptr, 0, 1.f, 0.f, 0, BATCHQ, 3 * HIDQ, HIDQ);
        k_gru<<<gb((long long)BATCHQ * HIDQ), 256, 0, stream>>>(gi, ghb, hbuf, hh);
        launch_gemm(stream, hh, HIDQ, Wouth, HIDQ, gout, GOUTQ, nullptr, 0,
                    b_out, nullptr, 0, 1.f, 0.f, 0, BATCHQ, GOUTQ, HIDQ);
        k_finalize<<<BATCHQ, 256, 0, stream>>>(gout, s, sN, lam, lamN, out_fp + it * BATCHQ);
    }

    // ---- outputs ----
    k_copy_xi<<<gb((long long)BATCHQ * NVARQ), 256, 0, stream>>>(sol, out_xi);
    k_avg<<<gb(BATCHQ), 256, 0, stream>>>(out_fp, out_primal, out_avg_fp, out_avg_pr);
}
